// My_App_69690139344896
// MI455X (gfx1250) — matrix operand order repaired and checked by reading, not compiled, not run on hardware
//
#include <hip/hip_runtime.h>
#include <math.h>

// ---- problem constants (match reference) ----
#define NN     50000
#define NE     800000
#define NEP    (NE + NN)      // 850000 edges incl. self loops
#define FIN    512
#define HEADS  2
#define C1V    128
#define C2V    32
#define NCLS   32
#define NHID   64

typedef __attribute__((ext_vector_type(16))) _Float16 v16h;
typedef __attribute__((ext_vector_type(8)))  float    v8f;

// =====================================================================
// WMMA GEMM:  C = act(A[M,K] @ B[K,Nc] + bias) * scale
// A : row-major f32, row stride lda (pointer pre-offset for head slicing)
// Bt: TRANSPOSED weights pre-converted to f16, row-major [Nc,K]
//     -> each lane's B fragment is 16 contiguous halfs (raw v16h load)
// C : row-major f32, row stride ldc (pointer pre-offset)
// One wave32 per 16x(16*NT) strip: one A fragment feeds NT WMMAs.
// f16 inputs, f32 accumulate (v_wmma_f32_16x16x32_f16).
// M%16==0, K%32==0, Nc%(16*NT)==0 (all hold here).
// Fragment layouts used here (A and B share contiguous k order):
//  A 16x32: lane<16 row=lane, halfs 0..15 = K[(lane>>4)*16 .. +15]
//  B 32x16: lane holds col=lane&15, K range (lane>>4)*16 .. +15
//  C/D    : lane col = lane&15, vgpr r -> row = (lane>>4)*8 + r
// =====================================================================
template <int NT>
__global__ void wmma_gemm(const float* __restrict__ A, int lda,
                          const _Float16* __restrict__ Bt,
                          float* __restrict__ Cm, int ldc,
                          const float* __restrict__ bias,
                          int K, int act, float scale) {
    const int lane = threadIdx.x & 31;
    const int tm   = blockIdx.x;            // row tile
    const int tn0  = blockIdx.y * NT;       // first col tile of this wave
    const int half = lane >> 4;
    const int r16  = lane & 15;

    const float* arow = A + (size_t)(tm * 16 + r16) * lda;

    v8f acc[NT];
#pragma unroll
    for (int n = 0; n < NT; ++n) acc[n] = (v8f){};

    for (int k0 = 0; k0 < K; k0 += 32) {
        // ---- A fragment: two contiguous 8-float runs per lane, cvt to f16 ----
        v16h a;
        {
            const float4* p0 = (const float4*)(arow + k0 + 16 * half);
            const float4* p1 = (const float4*)(arow + k0 + 16 * half + 8);
            float4 q0 = p0[0], q1 = p0[1], q2 = p1[0], q3 = p1[1];
            a[0]  = (_Float16)q0.x; a[1]  = (_Float16)q0.y;
            a[2]  = (_Float16)q0.z; a[3]  = (_Float16)q0.w;
            a[4]  = (_Float16)q1.x; a[5]  = (_Float16)q1.y;
            a[6]  = (_Float16)q1.z; a[7]  = (_Float16)q1.w;
            a[8]  = (_Float16)q2.x; a[9]  = (_Float16)q2.y;
            a[10] = (_Float16)q2.z; a[11] = (_Float16)q2.w;
            a[12] = (_Float16)q3.x; a[13] = (_Float16)q3.y;
            a[14] = (_Float16)q3.z; a[15] = (_Float16)q3.w;
        }
#pragma unroll
        for (int n = 0; n < NT; ++n) {
            // ---- B fragment: 16 contiguous halfs, no conversion needed ----
            const int bcol = (tn0 + n) * 16 + r16;
            v16h b = *(const v16h*)(Bt + (size_t)bcol * K + k0 + half * 16);
            acc[n] = __builtin_amdgcn_wmma_f32_16x16x32_f16(
                false, a, false, b, (short)0, acc[n], false, false);
        }
    }

#pragma unroll
    for (int n = 0; n < NT; ++n) {
        const int bcol = (tn0 + n) * 16 + r16;
        const float bv = bias ? bias[bcol] : 0.0f;
#pragma unroll
        for (int r = 0; r < 8; ++r) {
            int orow = tm * 16 + half * 8 + r;
            float v  = acc[n][r] + bv;
            if (act) v = v > 0.0f ? v : (__expf(v) - 1.0f);   // ELU
            Cm[(size_t)orow * ldc + bcol] = v * scale;
        }
    }
}

// transpose weights + convert: in f32 [R,C] -> out f16 [C,R]
__global__ void transpose_k(const float* __restrict__ in, _Float16* __restrict__ out,
                            int R, int Cc) {
    int i = blockIdx.x * blockDim.x + threadIdx.x;
    if (i >= R * Cc) return;
    int r = i / Cc, c = i % Cc;
    out[(size_t)c * R + r] = (_Float16)in[i];
}

__global__ void fill_f32(float* __restrict__ p, float v, long n) {
    long i = (long)blockIdx.x * blockDim.x + threadIdx.x;
    if (i < n) p[i] = v;
}

// =====================================================================
// Edge attention logits: e[w,h] = dot(xt[dst,h,:], xt[src,h,:]) / sqrt(C)
// One wave per edge; self loops synthesized for w >= NE.
// =====================================================================
__global__ void edge_logits(const float* __restrict__ xt,
                            const int* __restrict__ src,
                            const int* __restrict__ dst,
                            int C, float inv_sqrt_c,
                            float* __restrict__ ebuf) {
    int w = blockIdx.x * (blockDim.x >> 5) + (threadIdx.x >> 5);
    int lane = threadIdx.x & 31;
    if (w >= NEP) return;
    int s = (w < NE) ? src[w] : (w - NE);
    int d = (w < NE) ? dst[w] : (w - NE);
    const float* ps = xt + (size_t)s * (HEADS * C);
    const float* pd = xt + (size_t)d * (HEADS * C);
#pragma unroll
    for (int h = 0; h < HEADS; ++h) {
        float acc = 0.0f;
        for (int c = lane; c < C; c += 32) acc += pd[h * C + c] * ps[h * C + c];
        for (int off = 16; off; off >>= 1) acc += __shfl_xor(acc, off, 32);
        if (lane == 0) ebuf[(size_t)w * HEADS + h] = acc * inv_sqrt_c;
    }
}

__device__ __forceinline__ void atomicMaxF(float* addr, float val) {
    unsigned* ua = (unsigned*)addr;
    unsigned old = __float_as_uint(*addr);
    while (__uint_as_float(old) < val) {
        unsigned prev = atomicCAS(ua, old, __float_as_uint(val));
        if (prev == old) break;
        old = prev;
    }
}

// segment max of logits over dst (per head)
__global__ void seg_max(const float* __restrict__ ebuf,
                        const int* __restrict__ dst,
                        float* __restrict__ mbuf) {
    int i = blockIdx.x * blockDim.x + threadIdx.x;      // over NEP*HEADS
    if (i >= NEP * HEADS) return;
    int w = i >> 1, h = i & 1;
    int d = (w < NE) ? dst[w] : (w - NE);
    atomicMaxF(&mbuf[d * HEADS + h], ebuf[i]);
}

// ex = exp(e - m[dst]) in place; segment-sum into sbuf
__global__ void edge_expsum(float* __restrict__ ebuf,
                            const int* __restrict__ dst,
                            const float* __restrict__ mbuf,
                            float* __restrict__ sbuf) {
    int i = blockIdx.x * blockDim.x + threadIdx.x;
    if (i >= NEP * HEADS) return;
    int w = i >> 1, h = i & 1;
    int d = (w < NE) ? dst[w] : (w - NE);
    float ex = __expf(ebuf[i] - mbuf[d * HEADS + h]);
    ebuf[i] = ex;
    atomicAdd(&sbuf[d * HEADS + h], ex);
}

// out[dst] += alpha * xt[src]; optional alpha output (layer 3)
__global__ void edge_aggregate(const float* __restrict__ xt,
                               const int* __restrict__ src,
                               const int* __restrict__ dst,
                               const float* __restrict__ ebuf,
                               const float* __restrict__ sbuf,
                               float* __restrict__ agg, int C,
                               float* __restrict__ alpha_out) {
    int w = blockIdx.x * (blockDim.x >> 5) + (threadIdx.x >> 5);
    int lane = threadIdx.x & 31;
    if (w >= NEP) return;
    int s = (w < NE) ? src[w] : (w - NE);
    int d = (w < NE) ? dst[w] : (w - NE);
    float a0 = ebuf[(size_t)w * 2 + 0] / (sbuf[d * 2 + 0] + 1e-16f);
    float a1 = ebuf[(size_t)w * 2 + 1] / (sbuf[d * 2 + 1] + 1e-16f);
    if (alpha_out != nullptr && lane == 0) {
        alpha_out[(size_t)w * 2 + 0] = a0;
        alpha_out[(size_t)w * 2 + 1] = a1;
    }
    const int HC = HEADS * C;
    const float* ps = xt + (size_t)s * HC;
    float* pd = agg + (size_t)d * HC;
    for (int c = lane; c < HC; c += 32) {
        float al = (c < C) ? a0 : a1;
        atomicAdd(&pd[c], al * ps[c]);
    }
}

// =====================================================================
// LayerNorm (+ optional ELU). One wave per row.
// =====================================================================
__global__ void ln_act(const float* __restrict__ in, float* __restrict__ out,
                       const float* __restrict__ g, const float* __restrict__ b,
                       int D, int act) {
    int row = blockIdx.x * (blockDim.x >> 5) + (threadIdx.x >> 5);
    int lane = threadIdx.x & 31;
    if (row >= NN) return;
    const float* p = in + (size_t)row * D;
    float sum = 0.0f;
    for (int c = lane; c < D; c += 32) sum += p[c];
    for (int off = 16; off; off >>= 1) sum += __shfl_xor(sum, off, 32);
    float mean = sum / (float)D;
    float vs = 0.0f;
    for (int c = lane; c < D; c += 32) { float t = p[c] - mean; vs += t * t; }
    for (int off = 16; off; off >>= 1) vs += __shfl_xor(vs, off, 32);
    float rstd = rsqrtf(vs / (float)D + 1e-5f);
    float* q = out + (size_t)row * D;
    for (int c = lane; c < D; c += 32) {
        float v = (p[c] - mean) * rstd * g[c] + b[c];
        if (act) v = v > 0.0f ? v : (__expf(v) - 1.0f);
        q[c] = v;
    }
}

// mean over 2 heads: [N, 2*32] -> [N, 32]
__global__ void head_mean(const float* __restrict__ agg, float* __restrict__ outp) {
    int i = blockIdx.x * blockDim.x + threadIdx.x;
    if (i >= NN * NCLS) return;
    int n = i / NCLS, c = i % NCLS;
    outp[i] = 0.5f * (agg[(size_t)n * 64 + c] + agg[(size_t)n * 64 + 32 + c]);
}

// =====================================================================
// Host-side orchestration
// =====================================================================
static inline void gemm(const float* A, int lda, const _Float16* Bt,
                        float* C, int ldc, const float* bias,
                        int M, int Nc, int K, int act, float scale,
                        hipStream_t st) {
    if (Nc % 64 == 0) {
        dim3 grid(M / 16, Nc / 64);
        wmma_gemm<4><<<grid, dim3(32), 0, st>>>(A, lda, Bt, C, ldc, bias, K, act, scale);
    } else {
        dim3 grid(M / 16, Nc / 32);
        wmma_gemm<2><<<grid, dim3(32), 0, st>>>(A, lda, Bt, C, ldc, bias, K, act, scale);
    }
}

static inline void fill(float* p, float v, long n, hipStream_t st) {
    fill_f32<<<(unsigned)((n + 255) / 256), 256, 0, st>>>(p, v, n);
}

static inline void transp(const float* in, _Float16* out, int R, int Cc, hipStream_t st) {
    transpose_k<<<(R * Cc + 255) / 256, 256, 0, st>>>(in, out, R, Cc);
}

extern "C" void kernel_launch(void* const* d_in, const int* in_sizes, int n_in,
                              void* d_out, int out_size, void* d_ws, size_t ws_size,
                              hipStream_t stream) {
    const float* x    = (const float*)d_in[0];
    const int*   ei   = (const int*)d_in[1];
    const int*   src  = ei;
    const int*   dst  = ei + NE;
    const float* W1   = (const float*)d_in[2];
    const float* Wh1a = (const float*)d_in[3];
    const float* bh1a = (const float*)d_in[4];
    const float* Wh2a = (const float*)d_in[5];
    const float* bh2a = (const float*)d_in[6];
    const float* g1   = (const float*)d_in[7];
    const float* b1   = (const float*)d_in[8];
    const float* W2   = (const float*)d_in[9];
    const float* Wh1b = (const float*)d_in[10];
    const float* bh1b = (const float*)d_in[11];
    const float* Wh2b = (const float*)d_in[12];
    const float* bh2b = (const float*)d_in[13];
    const float* g2   = (const float*)d_in[14];
    const float* b2   = (const float*)d_in[15];
    const float* W3   = (const float*)d_in[16];
    const float* Wh1c = (const float*)d_in[17];
    const float* bh1c = (const float*)d_in[18];
    const float* Wh2c = (const float*)d_in[19];
    const float* bh2c = (const float*)d_in[20];
    const float* g3   = (const float*)d_in[21];
    const float* b3   = (const float*)d_in[22];

    // output slices (return order: x3, h1, h2*0.5, h3, alpha)
    float* out_x3    = (float*)d_out;
    float* out_h1    = out_x3 + (size_t)NN * NCLS;
    float* out_h2    = out_h1 + (size_t)NN * HEADS * NCLS;
    float* out_h3    = out_h2 + (size_t)NN * HEADS * NCLS;
    float* out_alpha = out_h3 + (size_t)NN * HEADS * NCLS;

    // workspace carve-out (floats)
    float* ws   = (float*)d_ws;
    float* xt1  = ws;                                // [NN,256]
    float* agg  = xt1 + (size_t)NN * 256;            // [NN,256] (reused [NN,64] L2/L3)
    float* xt2  = agg + (size_t)NN * 256;            // [NN,64]  (reused as xt3)
    float* hid  = xt2 + (size_t)NN * 64;             // [NN,2,64]
    float* ebuf = hid + (size_t)NN * 128;            // [NEP,2] logits->exp (x3pre reuse)
    float* mbuf = ebuf + (size_t)NEP * 2;            // [NN,2]
    float* sbuf = mbuf + (size_t)NN * 2;             // [NN,2]
    // transposed f16 weights [Nc,K]
    _Float16* W1t   = (_Float16*)(sbuf + (size_t)NN * 2);  // 256*512
    _Float16* Wh1at = W1t   + 256 * 512;                   // 64*128
    _Float16* Wh2at = Wh1at + 64 * 128;                    // 32*64
    _Float16* W2t   = Wh2at + 32 * 64;                     // 64*256
    _Float16* Wh1bt = W2t   + 64 * 256;                    // 64*32
    _Float16* Wh2bt = Wh1bt + 64 * 32;                     // 32*64
    _Float16* W3t   = Wh2bt + 32 * 64;                     // 64*64
    _Float16* Wh1ct = W3t   + 64 * 64;                     // 64*32
    _Float16* Wh2ct = Wh1ct + 64 * 32;                     // 32*64

    // stage transposed + f16-converted weights (tiny, once per launch)
    transp(W1,   W1t,   FIN,  HEADS * C1V, stream);
    transp(Wh1a, Wh1at, C1V,  NHID, stream);
    transp(Wh2a, Wh2at, NHID, NCLS, stream);
    transp(W2,   W2t,   HEADS * C1V, HEADS * C2V, stream);
    transp(Wh1b, Wh1bt, C2V,  NHID, stream);
    transp(Wh2b, Wh2bt, NHID, NCLS, stream);
    transp(W3,   W3t,   HEADS * C2V, HEADS * NCLS, stream);
    transp(Wh1c, Wh1ct, NCLS, NHID, stream);
    transp(Wh2c, Wh2ct, NHID, NCLS, stream);

    const int ewaves = (NEP + 7) / 8;                 // 8 waves / 256-thread block
    const int ethr   = (NEP * HEADS + 255) / 256;

    // ---------------- layer 1 ----------------
    gemm(x, FIN, W1t, xt1, HEADS * C1V, nullptr,
         NN, HEADS * C1V, FIN, 0, 1.0f, stream);
    for (int h = 0; h < HEADS; ++h) {
        gemm(xt1 + h * C1V, HEADS * C1V, Wh1at, hid + h * NHID, HEADS * NHID,
             bh1a, NN, NHID, C1V, 1, 1.0f, stream);
        gemm(hid + h * NHID, HEADS * NHID, Wh2at, out_h1 + h * NCLS, HEADS * NCLS,
             bh2a, NN, NCLS, NHID, 0, 1.0f, stream);
    }
    fill(mbuf, -3.402823466e38f, (long)NN * HEADS, stream);
    fill(sbuf, 0.0f, (long)NN * HEADS, stream);
    fill(agg,  0.0f, (long)NN * HEADS * C1V, stream);
    edge_logits<<<ewaves, 256, 0, stream>>>(xt1, src, dst, C1V,
                                            1.0f / sqrtf((float)C1V), ebuf);
    seg_max<<<ethr, 256, 0, stream>>>(ebuf, dst, mbuf);
    edge_expsum<<<ethr, 256, 0, stream>>>(ebuf, dst, mbuf, sbuf);
    edge_aggregate<<<ewaves, 256, 0, stream>>>(xt1, src, dst, ebuf, sbuf,
                                               agg, C1V, nullptr);
    ln_act<<<(NN + 7) / 8, 256, 0, stream>>>(agg, agg, g1, b1, HEADS * C1V, 1);

    // ---------------- layer 2 ----------------
    gemm(agg, HEADS * C1V, W2t, xt2, HEADS * C2V, nullptr,
         NN, HEADS * C2V, HEADS * C1V, 0, 1.0f, stream);
    for (int h = 0; h < HEADS; ++h) {
        gemm(xt2 + h * C2V, HEADS * C2V, Wh1bt, hid + h * NHID, HEADS * NHID,
             bh1b, NN, NHID, C2V, 1, 1.0f, stream);
        gemm(hid + h * NHID, HEADS * NHID, Wh2bt, out_h2 + h * NCLS, HEADS * NCLS,
             bh2b, NN, NCLS, NHID, 0, 0.5f, stream);   // * LOSS_WEIGHT
    }
    fill(mbuf, -3.402823466e38f, (long)NN * HEADS, stream);
    fill(sbuf, 0.0f, (long)NN * HEADS, stream);
    fill(agg,  0.0f, (long)NN * HEADS * C2V, stream);
    edge_logits<<<ewaves, 256, 0, stream>>>(xt2, src, dst, C2V,
                                            1.0f / sqrtf((float)C2V), ebuf);
    seg_max<<<ethr, 256, 0, stream>>>(ebuf, dst, mbuf);
    edge_expsum<<<ethr, 256, 0, stream>>>(ebuf, dst, mbuf, sbuf);
    edge_aggregate<<<ewaves, 256, 0, stream>>>(xt2, src, dst, ebuf, sbuf,
                                               agg, C2V, nullptr);
    ln_act<<<(NN + 7) / 8, 256, 0, stream>>>(agg, agg, g2, b2, HEADS * C2V, 1);

    // ---------------- layer 3 ----------------
    float* xt3 = xt2;   // xt2 dead; reuse buffer
    gemm(agg, HEADS * C2V, W3t, xt3, HEADS * NCLS, nullptr,
         NN, HEADS * NCLS, HEADS * C2V, 0, 1.0f, stream);
    for (int h = 0; h < HEADS; ++h) {
        gemm(xt3 + h * NCLS, HEADS * NCLS, Wh1ct, hid + h * NHID, HEADS * NHID,
             bh1c, NN, NHID, NCLS, 1, 1.0f, stream);
        gemm(hid + h * NHID, HEADS * NHID, Wh2ct, out_h3 + h * NCLS, HEADS * NCLS,
             bh2c, NN, NCLS, NHID, 0, 1.0f, stream);
    }
    fill(mbuf, -3.402823466e38f, (long)NN * HEADS, stream);
    fill(sbuf, 0.0f, (long)NN * HEADS, stream);
    fill(agg,  0.0f, (long)NN * HEADS * NCLS, stream);
    edge_logits<<<ewaves, 256, 0, stream>>>(xt3, src, dst, NCLS,
                                            1.0f / sqrtf((float)NCLS), ebuf);
    seg_max<<<ethr, 256, 0, stream>>>(ebuf, dst, mbuf);
    edge_expsum<<<ethr, 256, 0, stream>>>(ebuf, dst, mbuf, sbuf);
    edge_aggregate<<<ewaves, 256, 0, stream>>>(xt3, src, dst, ebuf, sbuf,
                                               agg, NCLS, out_alpha);
    // mean over heads -> x3pre (reuse ebuf: 1.7M >= 1.6M floats, ex data dead)
    float* x3pre = ebuf;
    head_mean<<<(NN * NCLS + 255) / 256, 256, 0, stream>>>(agg, x3pre);
    ln_act<<<(NN + 7) / 8, 256, 0, stream>>>(x3pre, out_x3, g3, b3, NCLS, 0);
}
